// MultiModalBehavioralGNN_4870492914137
// MI455X (gfx1250) — matrix operand order repaired and checked by reading, not compiled, not run on hardware
//
#include <hip/hip_runtime.h>
#include <hip/hip_bf16.h>
#include <math.h>

// Problem constants (match reference)
#define NND 20000          // nodes
#define NED 320000         // edges before self loops
#define ETOT (NED + NND)   // edges + self loops
#define NFIN 32
#define NH 4
#define NC 128
#define NL 3
#define NB 64
#define NEG_SLOPE 0.2f

typedef __attribute__((ext_vector_type(16))) __bf16 v16bf;
typedef __attribute__((ext_vector_type(8)))  __bf16 v8bf;
typedef __attribute__((ext_vector_type(8)))  float  v8f;

__device__ __forceinline__ __bf16 f2bf(float f) {
  unsigned u = __builtin_bit_cast(unsigned, f);
  u += 0x7FFFu + ((u >> 16) & 1u);          // round-to-nearest-even
  unsigned short h = (unsigned short)(u >> 16);
  return __builtin_bit_cast(__bf16, h);
}

__device__ __forceinline__ void atomicMaxF(float* addr, float val) {
  // signed-max for >=0, unsigned-min for <0; works with -inf init
  if (val >= 0.0f) atomicMax((int*)addr, __float_as_int(val));
  else             atomicMin((unsigned int*)addr, __float_as_uint(val));
}

__device__ __forceinline__ void edge_sd(const int* __restrict__ ei, int e, int& s, int& d) {
  if (e < NED) { s = ei[e]; d = ei[NED + e]; }
  else         { s = e - NED; d = e - NED; }   // self loop
}

__device__ __forceinline__ float lrelu(float v) {
  return v > 0.0f ? v : NEG_SLOPE * v;
}

// fp32 -> bf16 bulk convert (n % 4 == 0)
__global__ void conv_bf16_k(const float* __restrict__ src, __bf16* __restrict__ dst, int n) {
  int i = 4 * (blockIdx.x * blockDim.x + threadIdx.x);
  if (i >= n) return;
  float4 v = *(const float4*)(src + i);
  dst[i + 0] = f2bf(v.x);
  dst[i + 1] = f2bf(v.y);
  dst[i + 2] = f2bf(v.z);
  dst[i + 3] = f2bf(v.w);
}

// fp32 B[K,N] -> bf16 Bt[N,K] (convert + transpose, weights are tiny/L2-resident)
__global__ void conv_tr_bf16_k(const float* __restrict__ B, __bf16* __restrict__ Bt,
                               int K, int N) {
  int i = blockIdx.x * blockDim.x + threadIdx.x;
  if (i >= N * K) return;
  int n = i / K, k = i - n * K;
  Bt[i] = f2bf(B[(size_t)k * N + n]);
}

// ---------------------------------------------------------------------------
// bf16 WMMA GEMM, f32 accumulate.  C[M,Nn] = act(A[M,K] @ B[K,Nn] + bias).
// A: bf16 [M,K] row-major.  Bt: bf16 [Nn,K] (transposed).  M%16==0, Nn%32==0,
// K%32==0.  One 16x32 output tile per wave (2 WMMAs sharing the A fragment),
// 8 waves per block.  Inner loop = 6 x b128 loads + 2 x v_wmma.
// ---------------------------------------------------------------------------
__global__ void gemm_bf16_wmma(const __bf16* __restrict__ A, const __bf16* __restrict__ Bt,
                               float* __restrict__ C, const float* __restrict__ bias,
                               int M, int Nn, int K, int act) {
  const int lane = threadIdx.x & 31;
  const int wave = threadIdx.x >> 5;
  const int tilesN = Nn >> 5;                  // 32-wide N tiles
  const int tilesTotal = (M >> 4) * tilesN;
  const int tile = blockIdx.x * (blockDim.x >> 5) + wave;
  if (tile >= tilesTotal) return;              // wave-uniform: EXEC stays all-1s
  const int tm = tile / tilesN;
  const int tn = tile - tm * tilesN;
  const int half = lane >> 4;                  // 0: lanes 0-15, 1: lanes 16-31
  const int l15  = lane & 15;

  const int rowA = tm * 16 + l15;
  const int col0 = tn * 32 + l15;
  const int col1 = col0 + 16;
  v8f acc0 = {}, acc1 = {};

  const __bf16* arow = A  + (size_t)rowA * K;
  const __bf16* b0r  = Bt + (size_t)col0 * K;
  const __bf16* b1r  = Bt + (size_t)col1 * K;

  for (int k0 = 0; k0 < K; k0 += 32) {
    // A fragment (contiguous K order, matching B):
    //  lanes 0-15 : elems 0..15 -> K=k0+0..15
    //  lanes 16-31: elems 0..15 -> K=k0+16..31
    const __bf16* ap = arow + k0 + half * 16;
    v8bf a0 = *(const v8bf*)(ap);
    v8bf a1 = *(const v8bf*)(ap + 8);
    v16bf av = __builtin_shufflevector(a0, a1, 0,1,2,3,4,5,6,7,8,9,10,11,12,13,14,15);

    // B fragment: lanes 0-15 hold K=k0+0..15, lanes 16-31 hold K=k0+16..31
    const int kb = k0 + half * 16;
    v8bf c0 = *(const v8bf*)(b0r + kb);
    v8bf c1 = *(const v8bf*)(b0r + kb + 8);
    v16bf bv0 = __builtin_shufflevector(c0, c1, 0,1,2,3,4,5,6,7,8,9,10,11,12,13,14,15);
    v8bf d0 = *(const v8bf*)(b1r + kb);
    v8bf d1 = *(const v8bf*)(b1r + kb + 8);
    v16bf bv1 = __builtin_shufflevector(d0, d1, 0,1,2,3,4,5,6,7,8,9,10,11,12,13,14,15);

    acc0 = __builtin_amdgcn_wmma_f32_16x16x32_bf16(false, av, false, bv0,
                                                   (short)0, acc0, false, false);
    acc1 = __builtin_amdgcn_wmma_f32_16x16x32_bf16(false, av, false, bv1,
                                                   (short)0, acc1, false, false);
  }

  const float bv0s = bias ? bias[col0] : 0.0f;
  const float bv1s = bias ? bias[col1] : 0.0f;
  const int rbase = tm * 16 + half * 8;        // f32 C/D layout: VGPR v -> M = v (+8)
#pragma unroll
  for (int v = 0; v < 8; ++v) {
    float r0 = acc0[v] + bv0s;
    float r1 = acc1[v] + bv1s;
    if (act == 1) { r0 = fmaxf(r0, 0.0f); r1 = fmaxf(r1, 0.0f); }
    C[(size_t)(rbase + v) * Nn + col0] = r0;
    C[(size_t)(rbase + v) * Nn + col1] = r1;
  }
}

// a_s[n,h] = <xh[n,h,:], att_src[h,:]>, a_d likewise. One thread per (n,h).
__global__ void att_scores(const float* __restrict__ xh,
                           const float* __restrict__ attS, const float* __restrict__ attD,
                           float* __restrict__ as_, float* __restrict__ ad_) {
  int idx = blockIdx.x * blockDim.x + threadIdx.x;
  if (idx >= NND * NH) return;
  int n = idx >> 2, h = idx & 3;
  const float* xp = xh + (size_t)n * (NH * NC) + h * NC;
  const float* sp = attS + h * NC;
  const float* dp = attD + h * NC;
  float ss = 0.f, dd = 0.f;
#pragma unroll 4
  for (int c = 0; c < NC; ++c) { float v = xp[c]; ss += v * sp[c]; dd += v * dp[c]; }
  as_[idx] = ss; ad_[idx] = dd;
}

__global__ void fill_f(float* __restrict__ p, float v, int n) {
  int i = blockIdx.x * blockDim.x + threadIdx.x;
  if (i < n) p[i] = v;
}

// p[n*NC + c] = bias[c]  (initialize accumulator with layer bias)
__global__ void fill_bias_rows(float* __restrict__ p, const float* __restrict__ bias, int total) {
  int i = blockIdx.x * blockDim.x + threadIdx.x;
  if (i < total) p[i] = bias[i & (NC - 1)];
}

__global__ void edge_max_k(const int* __restrict__ ei, const float* __restrict__ as_,
                           const float* __restrict__ ad_, float* __restrict__ mx) {
  int idx = blockIdx.x * blockDim.x + threadIdx.x;
  if (idx >= ETOT * NH) return;
  int e = idx >> 2, h = idx & 3;
  int s, d; edge_sd(ei, e, s, d);
  float v = lrelu(as_[s * NH + h] + ad_[d * NH + h]);
  atomicMaxF(&mx[d * NH + h], v);
}

__global__ void edge_exp_k(const int* __restrict__ ei, const float* __restrict__ as_,
                           const float* __restrict__ ad_, const float* __restrict__ mx,
                           float* __restrict__ ex, float* __restrict__ den) {
  int idx = blockIdx.x * blockDim.x + threadIdx.x;
  if (idx >= ETOT * NH) return;
  int e = idx >> 2, h = idx & 3;
  int s, d; edge_sd(ei, e, s, d);
  float v = lrelu(as_[s * NH + h] + ad_[d * NH + h]);
  float ev = __expf(v - mx[d * NH + h]);
  ex[idx] = ev;
  unsafeAtomicAdd(&den[d * NH + h], ev);
}

// One block (128 threads = channels) per edge.  Head mean folded in:
// hout[dst,c] += (1/H) * sum_h alpha[e,h] * xh[src,h,c]   -> 1 atomic per channel
__global__ void edge_agg_k(const int* __restrict__ ei, const float* __restrict__ xh,
                           const float* __restrict__ ex, const float* __restrict__ den,
                           float* __restrict__ hout) {
  __shared__ float al[NH];
  int e = blockIdx.x;
  int s, d; edge_sd(ei, e, s, d);
  if (threadIdx.x < NH)
    al[threadIdx.x] = ex[(size_t)e * NH + threadIdx.x] /
                      den[(size_t)d * NH + threadIdx.x] * (1.0f / NH);
  __syncthreads();
  int c = threadIdx.x;
  const float* xp = xh + (size_t)s * (NH * NC);
  float v = al[0] * xp[c] + al[1] * xp[NC + c] + al[2] * xp[2 * NC + c] + al[3] * xp[3 * NC + c];
  unsafeAtomicAdd(&hout[(size_t)d * NC + c], v);
}

// global mean pool (accumulate), one block per node, 128 channel threads
__global__ void pool_accum_k(const float* __restrict__ h, const int* __restrict__ batch,
                             float* __restrict__ g, float* __restrict__ cnt) {
  int n = blockIdx.x;
  int b = batch[n];
  if (threadIdx.x == 0) unsafeAtomicAdd(&cnt[b], 1.0f);
  unsafeAtomicAdd(&g[(size_t)b * NC + threadIdx.x], h[(size_t)n * NC + threadIdx.x]);
}

__global__ void pool_div_k(float* __restrict__ g, const float* __restrict__ cnt, int total) {
  int i = blockIdx.x * blockDim.x + threadIdx.x;
  if (i < total) g[i] /= fmaxf(cnt[i / NC], 1.0f);
}

// anomaly = sigmoid(t1 @ a_W2 + a_b2), N=1 so scalar (64 dots of length 64)
__global__ void anomaly_head_k(const float* __restrict__ t1, const float* __restrict__ W2,
                               const float* __restrict__ b2, float* __restrict__ out) {
  int b = threadIdx.x;
  if (b >= NB) return;
  float s = b2[0];
#pragma unroll 4
  for (int j = 0; j < NC / 2; ++j) s += t1[b * (NC / 2) + j] * W2[j];
  out[b] = 1.0f / (1.0f + __expf(-s));
}

extern "C" void kernel_launch(void* const* d_in, const int* in_sizes, int n_in,
                              void* d_out, int out_size, void* d_ws, size_t ws_size,
                              hipStream_t stream) {
  const float* x      = (const float*)d_in[0];
  const int*   ei     = (const int*)d_in[1];
  const int*   batch  = (const int*)d_in[3];
  const float* node_W = (const float*)d_in[4];
  const float* node_b = (const float*)d_in[5];
  const float* gat_W  = (const float*)d_in[6];
  const float* att_s  = (const float*)d_in[7];
  const float* att_d  = (const float*)d_in[8];
  const float* gat_b  = (const float*)d_in[9];
  const float* aW1 = (const float*)d_in[10];
  const float* ab1 = (const float*)d_in[11];
  const float* aW2 = (const float*)d_in[12];
  const float* ab2 = (const float*)d_in[13];
  const float* pW1 = (const float*)d_in[14];
  const float* pb1 = (const float*)d_in[15];
  const float* pW2 = (const float*)d_in[16];
  const float* pb2 = (const float*)d_in[17];

  float* out      = (float*)d_out;
  float* out_anom = out;                       // [64]
  float* out_pat  = out + NB;                  // [64,32]
  float* out_h    = out + NB + NB * 32;        // [N,C] final node features

  float* ws = (float*)d_ws;
  size_t o = 0;                                // offsets in floats (16B aligned)
  float* xh  = ws + o; o += (size_t)NND * NH * NC;   // [N,H,C]
  float* hA  = ws + o; o += (size_t)NND * NC;
  float* hB  = ws + o; o += (size_t)NND * NC;
  float* as_ = ws + o; o += (size_t)NND * NH;
  float* ad_ = ws + o; o += (size_t)NND * NH;
  float* mx  = ws + o; o += (size_t)NND * NH;
  float* den = ws + o; o += (size_t)NND * NH;
  float* ex  = ws + o; o += (size_t)ETOT * NH;
  float* g   = ws + o; o += (size_t)NB * NC;
  float* cnt = ws + o; o += NB;
  float* t1  = ws + o; o += NB * (NC / 2);
  float* t2  = ws + o; o += NB * (NC / 2);
  __bf16* Abf = (__bf16*)(ws + o); o += (size_t)NND * NC / 2;  // bf16 A copy (max M*K)
  __bf16* Btb = (__bf16*)(ws + o); o += (size_t)NC * NH * NC / 2; // bf16 B^T (max K*N)

  auto gemm = [&](const float* A, const float* Bm, float* Cm, const float* bias,
                  int M, int Nn, int K, int act) {
    int na = M * K, nb = Nn * K;
    conv_bf16_k<<<(na / 4 + 255) / 256, 256, 0, stream>>>(A, Abf, na);
    conv_tr_bf16_k<<<(nb + 255) / 256, 256, 0, stream>>>(Bm, Btb, K, Nn);
    int tiles  = (M / 16) * (Nn / 32);
    int blocks = (tiles + 7) / 8;               // 8 waves (tiles) per block
    gemm_bf16_wmma<<<blocks, 256, 0, stream>>>(Abf, Btb, Cm, bias, M, Nn, K, act);
  };

  const int NHthreads = NND * NH;
  const int EHthreads = ETOT * NH;

  // node embedding: hA = x @ node_W + node_b   [20000,32]@[32,128]
  gemm(x, node_W, hA, node_b, NND, NC, NFIN, 0);

  const float* hin = hA;
  for (int l = 0; l < NL; ++l) {
    float* hout = (l == 0) ? hB : ((l == 1) ? hA : out_h);

    // xh = hin @ W_l   [20000,128]@[128,512]   (dominant GEMM -> WMMA)
    gemm(hin, gat_W + (size_t)l * NC * NH * NC, xh, nullptr, NND, NH * NC, NC, 0);

    att_scores<<<(NHthreads + 255) / 256, 256, 0, stream>>>(
        xh, att_s + (size_t)l * NH * NC, att_d + (size_t)l * NH * NC, as_, ad_);

    fill_f<<<(NHthreads + 255) / 256, 256, 0, stream>>>(mx, -INFINITY, NHthreads);
    fill_f<<<(NHthreads + 255) / 256, 256, 0, stream>>>(den, 0.0f, NHthreads);

    edge_max_k<<<(EHthreads + 255) / 256, 256, 0, stream>>>(ei, as_, ad_, mx);
    edge_exp_k<<<(EHthreads + 255) / 256, 256, 0, stream>>>(ei, as_, ad_, mx, ex, den);

    fill_bias_rows<<<((NND * NC) + 255) / 256, 256, 0, stream>>>(hout, gat_b + l * NC, NND * NC);
    edge_agg_k<<<ETOT, NC, 0, stream>>>(ei, xh, ex, den, hout);

    hin = hout;
  }

  // global mean pool
  fill_f<<<((NB * NC) + 255) / 256, 256, 0, stream>>>(g, 0.0f, NB * NC);
  fill_f<<<1, 64, 0, stream>>>(cnt, 0.0f, NB);
  pool_accum_k<<<NND, NC, 0, stream>>>(hin, batch, g, cnt);
  pool_div_k<<<((NB * NC) + 255) / 256, 256, 0, stream>>>(g, cnt, NB * NC);

  // heads
  gemm(g, aW1, t1, ab1, NB, NC / 2, NC, 1);              // relu(g@a_W1+b1)  (WMMA)
  anomaly_head_k<<<1, 64, 0, stream>>>(t1, aW2, ab2, out_anom);
  gemm(g, pW1, t2, pb1, NB, NC / 2, NC, 1);              // relu(g@p_W1+b1)  (WMMA)
  gemm(t2, pW2, out_pat, pb2, NB, 32, NC / 2, 0);        // t2@p_W2+b2       (WMMA)
}
